// GPT_1211180777728
// MI455X (gfx1250) — matrix operand order repaired and checked by reading, not compiled, not run on hardware
//
#include <hip/hip_runtime.h>
#include <hip/hip_bf16.h>
#include <stdint.h>

// ---------------------------------------------------------------------------
// Types for CDNA5 WMMA (wave32): V_WMMA_F32_16X16X32_BF16
// ---------------------------------------------------------------------------
typedef __attribute__((ext_vector_type(16))) __bf16         v16bf;
typedef __attribute__((ext_vector_type(8)))  float          v8f;
typedef __attribute__((ext_vector_type(16))) unsigned short v16u;

union Frag16 { v16u u; v16bf b; };

#define GF_OUT_BF16 1
#define GF_RELU     2

#if defined(__has_builtin)
#if __has_builtin(__builtin_amdgcn_global_load_async_to_lds_b128) && \
    __has_builtin(__builtin_amdgcn_s_wait_asynccnt)
#define USE_ASYNC_COPY 1
#endif
#endif
#ifndef USE_ASYNC_COPY
#define USE_ASYNC_COPY 0
#endif

#if USE_ASYNC_COPY
// async-copy pointer types: global = addrspace(1), LDS = addrspace(3)
typedef int vint4_t __attribute__((vector_size(16)));
typedef __attribute__((address_space(1))) vint4_t* gv4p;
typedef __attribute__((address_space(3))) vint4_t* lv4p;
#endif

// round-to-nearest-even f32 -> bf16 (as raw u16)
__device__ __forceinline__ unsigned short f32_to_bf16(float f) {
    unsigned int u = __builtin_bit_cast(unsigned int, f);
    u += 0x7FFFu + ((u >> 16) & 1u);
    return (unsigned short)(u >> 16);
}
__device__ __forceinline__ unsigned int pack_bf16x2(float a, float b) {
    return (unsigned int)f32_to_bf16(a) | ((unsigned int)f32_to_bf16(b) << 16);
}

// reductions across the 16 lanes of a half-wave (lanes 0-15 / 16-31 stay separate)
__device__ __forceinline__ float hred_max(float v) {
    v = fmaxf(v, __shfl_xor(v, 1, 32));
    v = fmaxf(v, __shfl_xor(v, 2, 32));
    v = fmaxf(v, __shfl_xor(v, 4, 32));
    v = fmaxf(v, __shfl_xor(v, 8, 32));
    return v;
}
__device__ __forceinline__ float hred_sum(float v) {
    v += __shfl_xor(v, 1, 32);
    v += __shfl_xor(v, 2, 32);
    v += __shfl_xor(v, 4, 32);
    v += __shfl_xor(v, 8, 32);
    return v;
}

// ---------------------------------------------------------------------------
// Embedding: h[m, :] = tok_emb[x[m], :] + pos_emb[m % T, :]
// ---------------------------------------------------------------------------
__global__ void embed_kernel(const int* __restrict__ x,
                             const float* __restrict__ tok,
                             const float* __restrict__ pos,
                             float* __restrict__ h) {
    const int m = blockIdx.x;
    const int t = m & 2047;            // T = 2048
    const int id = x[m];
    for (int e = threadIdx.x; e < 1024; e += 256)
        h[(size_t)m * 1024 + e] = tok[(size_t)id * 1024 + e] + pos[(size_t)t * 1024 + e];
}

// ---------------------------------------------------------------------------
// LayerNorm (E=1024), one block of 256 threads per row; bf16 output for GEMM A
// ---------------------------------------------------------------------------
__device__ __forceinline__ float blk_sum(float v, float* red) {
    const int t = threadIdx.x;
    red[t] = v; __syncthreads();
    for (int s = 128; s > 0; s >>= 1) {
        if (t < s) red[t] += red[t + s];
        __syncthreads();
    }
    float r = red[0];
    __syncthreads();
    return r;
}

__global__ __launch_bounds__(256)
void ln_kernel(const float* __restrict__ x,
               const float* __restrict__ g,
               const float* __restrict__ be,
               unsigned short* __restrict__ out) {
    __shared__ float red[256];
    const int row = blockIdx.x;
    const int tid = threadIdx.x;
    const float* xr = x + (size_t)row * 1024;

    float v[4], s = 0.0f;
    #pragma unroll
    for (int i = 0; i < 4; ++i) { v[i] = xr[tid + i * 256]; s += v[i]; }
    s = blk_sum(s, red);
    const float mean = s * (1.0f / 1024.0f);

    float sq = 0.0f;
    #pragma unroll
    for (int i = 0; i < 4; ++i) { float d = v[i] - mean; sq += d * d; }
    sq = blk_sum(sq, red);
    const float rstd = rsqrtf(sq * (1.0f / 1024.0f) + 1e-5f);

    #pragma unroll
    for (int i = 0; i < 4; ++i) {
        const int e = tid + i * 256;
        out[(size_t)row * 1024 + e] = f32_to_bf16((v[i] - mean) * rstd * g[e] + be[e]);
    }
}

// ---------------------------------------------------------------------------
// Generic GEMM: C[M,N] = act(A[M,K](bf16) @ W[K,N](f32->bf16) + bias) (+res)
// Weight addressing: W[k, n] = Wbase[k*wks + (n/nblk)*blkstride + (n%nblk)]
// Tile: 64x64, BK=64 (two WMMA K-steps per stage). 128 threads = 4 waves;
// wave w computes rows [16w,16w+16) x 64 cols -> 8x v_wmma per stage.
// A tile staged via GLOBAL_LOAD_ASYNC_TO_LDS_B128 when available.
// W tile: lane-coalesced f32 loads, packed bf16x2, transposed b128 LDS stores.
// ---------------------------------------------------------------------------
__global__ __launch_bounds__(128)
void gemm_bf16_kernel(const unsigned short* __restrict__ A, int K,
                      const float* __restrict__ W, int wks, int nblk, long long blkstride,
                      const float* __restrict__ bias, const float* __restrict__ res,
                      void* __restrict__ outp, int flags, int N) {
    __shared__ unsigned short tA[64][72];   // [m][k]
    __shared__ unsigned short tB[64][72];   // [n][k]  (transposed)

    const int m0   = blockIdx.y * 64;
    const int n0   = blockIdx.x * 64;
    const int tid  = threadIdx.x;
    const int wave = tid >> 5;
    const int lane = tid & 31;
    const int fm   = lane & 15;             // fragment row/col within 16
    const int kbA  = (lane >> 4) * 16;      // A-frag K base (contiguous k = 16h + s)
    const int kbB  = (lane >> 4) * 16;      // B-frag K base (contiguous k = 16h + s, same as A)

    v8f acc[4];
    #pragma unroll
    for (int g = 0; g < 4; ++g)
        #pragma unroll
        for (int i = 0; i < 8; ++i) acc[g][i] = 0.0f;

    // loader roles
    const int arow = tid >> 1;              // 0..63
    const int acb  = (tid & 1) * 32;        // 0 or 32 (halves)
    const int bn   = tid & 63;              // output column within tile
    const int bkh  = (tid >> 6) * 32;       // k-run base: 0 or 32
    const int ng   = n0 + bn;
    const size_t wcol = (size_t)(ng / nblk) * blkstride + (size_t)(ng % nblk);
    const int ar = wave * 16 + fm;          // this lane's A-frag row

    for (int k0 = 0; k0 < K; k0 += 64) {
        __syncthreads();
        // ---- stage A (bf16): 64 bytes per thread, contiguous ----
        {
            const unsigned short* ag = A + (size_t)(m0 + arow) * K + k0 + acb;
            if (k0 + 64 < K) __builtin_prefetch(ag + 64, 0, 3);
#if USE_ASYNC_COPY
            #pragma unroll
            for (int i = 0; i < 4; ++i)
                __builtin_amdgcn_global_load_async_to_lds_b128(
                    (gv4p)(ag + 16 * i),
                    (lv4p)&tA[arow][acb + 16 * i],
                    0, 0);
#else
            #pragma unroll
            for (int i = 0; i < 4; ++i)
                *(uint4*)&tA[arow][acb + 16 * i] = *(const uint4*)(ag + 16 * i);
#endif
        }
        // ---- stage W (f32 -> bf16, transposed): one column, 32 k per thread ----
        {
            const float* wg = W + (size_t)(k0 + bkh) * wks + wcol;
            #pragma unroll
            for (int i = 0; i < 4; ++i) {
                uint4 pk;
                pk.x = pack_bf16x2(wg[(size_t)(8 * i + 0) * wks], wg[(size_t)(8 * i + 1) * wks]);
                pk.y = pack_bf16x2(wg[(size_t)(8 * i + 2) * wks], wg[(size_t)(8 * i + 3) * wks]);
                pk.z = pack_bf16x2(wg[(size_t)(8 * i + 4) * wks], wg[(size_t)(8 * i + 5) * wks]);
                pk.w = pack_bf16x2(wg[(size_t)(8 * i + 6) * wks], wg[(size_t)(8 * i + 7) * wks]);
                *(uint4*)&tB[bn][bkh + 8 * i] = pk;
            }
        }
#if USE_ASYNC_COPY
        __builtin_amdgcn_s_wait_asynccnt(0);
#endif
        __syncthreads();

        // ---- two WMMA K-steps over the staged 64-deep tile ----
        #pragma unroll
        for (int kk = 0; kk < 64; kk += 32) {
            Frag16 a;
            #pragma unroll
            for (int q = 0; q < 8; ++q) {
                a.u[q]     = tA[ar][kk + kbA + q];
                a.u[8 + q] = tA[ar][kk + kbA + 8 + q];
            }
            #pragma unroll
            for (int g = 0; g < 4; ++g) {
                Frag16 bfr;
                const int bnr = g * 16 + fm;
                #pragma unroll
                for (int q = 0; q < 16; ++q) bfr.u[q] = tB[bnr][kk + kbB + q];
                acc[g] = __builtin_amdgcn_wmma_f32_16x16x32_bf16(
                    false, a.b, false, bfr.b, (short)0, acc[g], false, false);
            }
        }
    }

    // epilogue: C/D layout -> VGPR r: lanes 0-15 => M=r, lanes 16-31 => M=r+8
    #pragma unroll
    for (int g = 0; g < 4; ++g) {
        #pragma unroll
        for (int r = 0; r < 8; ++r) {
            const int m = m0 + wave * 16 + r + ((lane >> 4) << 3);
            const int n = n0 + g * 16 + fm;
            float v = acc[g][r];
            if (bias) v += bias[n];
            if (flags & GF_RELU) v = fmaxf(v, 0.0f);
            if (res) v += res[(size_t)m * N + n];
            if (flags & GF_OUT_BF16)
                ((unsigned short*)outp)[(size_t)m * N + n] = f32_to_bf16(v);
            else
                ((float*)outp)[(size_t)m * N + n] = v;
        }
    }
}

// ---------------------------------------------------------------------------
// Fused causal attention (flash-style, online softmax).
// One wave per (b, h, 16-query tile). Q/K/V/O are bf16 [B*T, E] with head
// offset h*64.  scale = E^-0.5 = 1/32 (matches reference's C**-0.5).
// QK^T: 2x wmma (D=64).  P@V: per 32-key chunk, P transposed via LDS, 4x wmma.
// ---------------------------------------------------------------------------
__global__ __launch_bounds__(32)
void attn_kernel(const unsigned short* __restrict__ Q,
                 const unsigned short* __restrict__ Kc,
                 const unsigned short* __restrict__ Vc,
                 unsigned short* __restrict__ O) {
    constexpr int TT = 2048, EE = 1024;
    const float scale = 0.03125f;

    __shared__ unsigned short P[16][40];     // probs tile, rows x 32 chunk cols

    const int qt    = blockIdx.x;            // query tile
    const int hh    = blockIdx.y;
    const int bb    = blockIdx.z;
    const int lane  = threadIdx.x;
    const int qbase = qt * 16;
    const int fm    = lane & 15;
    const int half  = lane >> 4;
    const int kbA   = half * 16;
    const int kbB   = half * 16;
    const size_t base = (size_t)bb * TT * EE + (size_t)hh * 64;

    // Q A-fragments for dims [0,32) and [32,64)
    Frag16 aq0, aq1;
    {
        const unsigned short* qrow = Q + base + (size_t)(qbase + fm) * EE;
        #pragma unroll
        for (int q = 0; q < 8; ++q) {
            aq0.u[q]     = qrow[kbA + q];
            aq0.u[8 + q] = qrow[kbA + 8 + q];
            aq1.u[q]     = qrow[32 + kbA + q];
            aq1.u[8 + q] = qrow[32 + kbA + 8 + q];
        }
    }

    float mrun[8], lrun[8];
    v8f o[4];
    #pragma unroll
    for (int r = 0; r < 8; ++r) { mrun[r] = -1e30f; lrun[r] = 0.0f; }
    #pragma unroll
    for (int g = 0; g < 4; ++g)
        #pragma unroll
        for (int i = 0; i < 8; ++i) o[g][i] = 0.0f;

    for (int s0 = 0; s0 < qbase + 16; s0 += 32) {
        // ---- scores: two 16x16 tiles covering keys [s0, s0+32) ----
        v8f s[2];
        #pragma unroll
        for (int j = 0; j < 2; ++j) {
            #pragma unroll
            for (int i = 0; i < 8; ++i) s[j][i] = 0.0f;
            const unsigned short* krow = Kc + base + (size_t)(s0 + 16 * j + fm) * EE;
            Frag16 bk0, bk1;
            #pragma unroll
            for (int q = 0; q < 16; ++q) {
                bk0.u[q] = krow[kbB + q];
                bk1.u[q] = krow[32 + kbB + q];
            }
            s[j] = __builtin_amdgcn_wmma_f32_16x16x32_bf16(
                false, aq0.b, false, bk0.b, (short)0, s[j], false, false);
            s[j] = __builtin_amdgcn_wmma_f32_16x16x32_bf16(
                false, aq1.b, false, bk1.b, (short)0, s[j], false, false);
            #pragma unroll
            for (int r = 0; r < 8; ++r) {
                const int kt = s0 + 16 * j + fm;
                const int qrow_g = qbase + r + half * 8;
                s[j][r] = (kt <= qrow_g) ? s[j][r] * scale : -1e30f;
            }
        }

        // ---- online softmax update over this 32-key chunk ----
        __syncthreads();
        #pragma unroll
        for (int r = 0; r < 8; ++r) {
            float cmax = hred_max(fmaxf(s[0][r], s[1][r]));
            const float nm  = fmaxf(mrun[r], cmax);
            const float fac = __expf(mrun[r] - nm);
            const float p0  = __expf(s[0][r] - nm);
            const float p1  = __expf(s[1][r] - nm);
            lrun[r] = lrun[r] * fac + hred_sum(p0 + p1);
            mrun[r] = nm;
            #pragma unroll
            for (int g = 0; g < 4; ++g) o[g][r] *= fac;
            const int rowM = r + half * 8;
            P[rowM][fm]      = f32_to_bf16(p0);
            P[rowM][16 + fm] = f32_to_bf16(p1);
        }
        __syncthreads();

        // ---- P (16x32) A-fragment from LDS ----
        Frag16 ap;
        #pragma unroll
        for (int q = 0; q < 8; ++q) {
            ap.u[q]     = P[fm][kbA + q];
            ap.u[8 + q] = P[fm][kbA + 8 + q];
        }
        // ---- accumulate O += P @ V_chunk (4 column groups of 16) ----
        #pragma unroll
        for (int g = 0; g < 4; ++g) {
            Frag16 bv;
            #pragma unroll
            for (int q = 0; q < 16; ++q) {
                const int key = s0 + kbB + q;
                bv.u[q] = Vc[base + (size_t)key * EE + g * 16 + fm];
            }
            o[g] = __builtin_amdgcn_wmma_f32_16x16x32_bf16(
                false, ap.b, false, bv.b, (short)0, o[g], false, false);
        }
    }

    // ---- normalize and write bf16 output into [B*T, E] with head offset ----
    #pragma unroll
    for (int r = 0; r < 8; ++r) {
        const float inv = 1.0f / lrun[r];
        const int rowM = r + half * 8;
        #pragma unroll
        for (int g = 0; g < 4; ++g)
            O[base + (size_t)(qbase + rowM) * EE + g * 16 + fm] =
                f32_to_bf16(o[g][r] * inv);
    }
}

// ---------------------------------------------------------------------------
// Orchestration
// ---------------------------------------------------------------------------
extern "C" void kernel_launch(void* const* d_in, const int* in_sizes, int n_in,
                              void* d_out, int out_size, void* d_ws, size_t ws_size,
                              hipStream_t stream) {
    (void)in_sizes; (void)n_in; (void)out_size; (void)ws_size;
    constexpr int Vv = 32000, Ee = 1024, Hh = 16, Ll = 8, Dd = 64,
                  Ff = 4096, Bb = 2, Tt = 2048, Mm = Bb * Tt;

    const int*   x    = (const int*)  d_in[0];
    const float* tok  = (const float*)d_in[1];
    const float* pos  = (const float*)d_in[2];
    const float* wq   = (const float*)d_in[3];
    const float* wk   = (const float*)d_in[4];
    const float* wv   = (const float*)d_in[5];
    const float* wo   = (const float*)d_in[6];
    const float* bo   = (const float*)d_in[7];
    const float* w1   = (const float*)d_in[8];
    const float* b1   = (const float*)d_in[9];
    const float* w2   = (const float*)d_in[10];
    const float* b2   = (const float*)d_in[11];
    const float* ln1g = (const float*)d_in[12];
    const float* ln1b = (const float*)d_in[13];
    const float* ln2g = (const float*)d_in[14];
    const float* ln2b = (const float*)d_in[15];
    const float* lnfg = (const float*)d_in[16];
    const float* lnfb = (const float*)d_in[17];
    const float* fcw  = (const float*)d_in[18];
    const float* fcb  = (const float*)d_in[19];

    char* p = (char*)d_ws;
    float*          h     = (float*)p;          p += (size_t)Mm * Ee * 4;
    unsigned short* xnb   = (unsigned short*)p; p += (size_t)Mm * Ee * 2;
    unsigned short* qb    = (unsigned short*)p; p += (size_t)Mm * Ee * 2;
    unsigned short* kb    = (unsigned short*)p; p += (size_t)Mm * Ee * 2;
    unsigned short* vb    = (unsigned short*)p; p += (size_t)Mm * Ee * 2;
    unsigned short* attnb = (unsigned short*)p; p += (size_t)Mm * Ee * 2;
    unsigned short* ffb   = (unsigned short*)p; p += (size_t)Mm * Ff * 2;

    embed_kernel<<<Mm, 256, 0, stream>>>(x, tok, pos, h);

    const dim3 gE(Ee / 64, Mm / 64);
    const dim3 gF(Ff / 64, Mm / 64);
    const dim3 gV(Vv / 64, Mm / 64);
    const dim3 gAttn(Tt / 16, Hh, Bb);

    for (int l = 0; l < Ll; ++l) {
        ln_kernel<<<Mm, 256, 0, stream>>>(h, ln1g + l * Ee, ln1b + l * Ee, xnb);
        // QKV: weight [H,E,D] -> wks=D, nblk=D, blkstride=E*D
        gemm_bf16_kernel<<<gE, 128, 0, stream>>>(xnb, Ee, wq + (size_t)l * Hh * Ee * Dd,
            Dd, Dd, (long long)Ee * Dd, nullptr, nullptr, qb, GF_OUT_BF16, Ee);
        gemm_bf16_kernel<<<gE, 128, 0, stream>>>(xnb, Ee, wk + (size_t)l * Hh * Ee * Dd,
            Dd, Dd, (long long)Ee * Dd, nullptr, nullptr, kb, GF_OUT_BF16, Ee);
        gemm_bf16_kernel<<<gE, 128, 0, stream>>>(xnb, Ee, wv + (size_t)l * Hh * Ee * Dd,
            Dd, Dd, (long long)Ee * Dd, nullptr, nullptr, vb, GF_OUT_BF16, Ee);

        attn_kernel<<<gAttn, 32, 0, stream>>>(qb, kb, vb, attnb);

        // h = h + attn @ wo + bo
        gemm_bf16_kernel<<<gE, 128, 0, stream>>>(attnb, Ee, wo + (size_t)l * Ee * Ee,
            Ee, Ee, 0, bo + l * Ee, h, h, 0, Ee);

        ln_kernel<<<Mm, 256, 0, stream>>>(h, ln2g + l * Ee, ln2b + l * Ee, xnb);
        // ff = relu(xn @ w1 + b1) -> bf16
        gemm_bf16_kernel<<<gF, 128, 0, stream>>>(xnb, Ee, w1 + (size_t)l * Ee * Ff,
            Ff, Ff, 0, b1 + l * Ff, nullptr, ffb, GF_OUT_BF16 | GF_RELU, Ff);
        // h = h + ff @ w2 + b2
        gemm_bf16_kernel<<<gE, 128, 0, stream>>>(ffb, Ff, w2 + (size_t)l * Ff * Ee,
            Ee, Ee, 0, b2 + l * Ee, h, h, 0, Ee);
    }

    ln_kernel<<<Mm, 256, 0, stream>>>(h, lnfg, lnfb, xnb);
    // logits = xn @ fc_w + fc_b  (f32 out to d_out)
    gemm_bf16_kernel<<<gV, 128, 0, stream>>>(xnb, Ee, fcw,
        Vv, Vv, 0, fcb, nullptr, d_out, 0, Vv);
}
